// Experts_80882824118673
// MI455X (gfx1250) — matrix operand order repaired and checked by reading, not compiled, not run on hardware
//
#include <hip/hip_runtime.h>

// MoE expert FFN for MI455X (gfx1250, wave32).
// Pass 0: one-time fp32->bf16 convert of x, and convert+transpose of W1/W2 so
//         both GEMM operands are K-contiguous bf16 (per-expert set ~109 MB -> L2 resident).
// Pass 1: inter = relu(x @ W1)   (bf16 -> ws)
// Pass 2: out   = inter @ W2     (fp32 -> d_out)
// GEMMs: 128x256 block tile, 8 wave32 in 2x4 (64x64 per wave), BK=64,
//        double-buffered LDS fed by global_load_async_to_lds_b128 (ASYNCcnt),
//        16 ds_load_b128 : 16 v_wmma per K-step per wave.
// Workspace assumption: ws_size >= ~832 MiB.

typedef __bf16 bf16_t;
typedef __bf16 bf16x4 __attribute__((ext_vector_type(4)));
typedef __bf16 bf16x8 __attribute__((ext_vector_type(8)));
typedef __bf16 v16bf  __attribute__((ext_vector_type(16)));
typedef float  v8f    __attribute__((ext_vector_type(8)));
typedef float  f32x4  __attribute__((ext_vector_type(4)));
typedef int    i32x4  __attribute__((ext_vector_type(4)));

// pointer-to-v4i in global (AS1) / LDS (AS3) address spaces, matching the
// async builtin's parameter types (confirmed by compiler diagnostics).
typedef __attribute__((address_space(1))) i32x4* g4p_t;
typedef __attribute__((address_space(3))) i32x4* l4p_t;

#define BM  128
#define BN  256
#define BK  64    // two 16x16x32 K-steps per stage
#define BKP 72    // padded LDS K-stride (144 B rows, 16B aligned)

#if defined(__gfx1250__) && __has_builtin(__builtin_amdgcn_global_load_async_to_lds_b128)
#define HAVE_ASYNC 1
#else
#define HAVE_ASYNC 0
#endif

#if HAVE_ASYNC
#if __has_builtin(__builtin_amdgcn_s_wait_asynccnt)
#define WAIT_ASYNC(n) __builtin_amdgcn_s_wait_asynccnt(n)
#else
#define WAIT_ASYNC(n) asm volatile("s_wait_asynccnt %0" ::"i"(n) : "memory")
#endif
#else
#define WAIT_ASYNC(n) ((void)0)
#endif

// 16-byte global -> LDS copy; async DMA (ASYNCcnt) on gfx1250.
__device__ __forceinline__ void cp16(const bf16_t* g, bf16_t* l) {
#if HAVE_ASYNC
  __builtin_amdgcn_global_load_async_to_lds_b128(
      (g4p_t)(unsigned long long)(size_t)g,
      (l4p_t)(unsigned int)(size_t)l,
      0, 0);
#else
  *reinterpret_cast<bf16x8*>(l) = *reinterpret_cast<const bf16x8*>(g);
#endif
}

// Assemble one 16x16x32 bf16 fragment from padded row-major-K LDS.
// A: row = m-base + (lane&15); chunks at K = ks+lh*8 and ks+16+lh*8
// B: row = n-base + (lane&15); chunks at K = ks+lh*8 and ks+16+lh*8
__device__ __forceinline__ v16bf frag_ld(const bf16_t* p, int row, int k0, int k1) {
  bf16x8 lo = *reinterpret_cast<const bf16x8*>(p + row * BKP + k0);
  bf16x8 hi = *reinterpret_cast<const bf16x8*>(p + row * BKP + k1);
  return __builtin_shufflevector(lo, hi, 0,1,2,3,4,5,6,7,8,9,10,11,12,13,14,15);
}

// ---------------- pass 0a: elementwise fp32 -> bf16 ----------------
__global__ __launch_bounds__(256)
void cast_f32_bf16(const float* __restrict__ in, bf16_t* __restrict__ out, size_t n) {
  size_t i = ((size_t)blockIdx.x * blockDim.x + threadIdx.x) * 4;
  size_t stride = (size_t)gridDim.x * blockDim.x * 4;
  for (; i < n; i += stride) {
    f32x4 d = *reinterpret_cast<const f32x4*>(in + i);
    bf16x4 t;
    t[0] = (bf16_t)d[0]; t[1] = (bf16_t)d[1];
    t[2] = (bf16_t)d[2]; t[3] = (bf16_t)d[3];
    *reinterpret_cast<bf16x4*>(out + i) = t;
  }
}

// ---------------- pass 0b: fp32 (R x C) -> bf16 transposed (C x R) ----------------
__global__ __launch_bounds__(256)
void transpose_cast_f32_bf16(const float* __restrict__ in, bf16_t* __restrict__ out,
                             int R, int C) {
  __shared__ bf16_t tile[32][36];
  const int e = blockIdx.z;
  in  += (size_t)e * R * C;
  out += (size_t)e * R * C;
  const int c0 = blockIdx.x * 32, r0 = blockIdx.y * 32;
  const int tid = threadIdx.x;
  const int r = tid >> 3, c4 = (tid & 7) * 4;

  f32x4 d = *reinterpret_cast<const f32x4*>(in + (size_t)(r0 + r) * C + c0 + c4);
  tile[r][c4 + 0] = (bf16_t)d[0];
  tile[r][c4 + 1] = (bf16_t)d[1];
  tile[r][c4 + 2] = (bf16_t)d[2];
  tile[r][c4 + 3] = (bf16_t)d[3];
  __syncthreads();

  bf16x4 o;
  o[0] = tile[c4 + 0][r]; o[1] = tile[c4 + 1][r];
  o[2] = tile[c4 + 2][r]; o[3] = tile[c4 + 3][r];
  *reinterpret_cast<bf16x4*>(out + (size_t)(c0 + r) * R + r0 + c4) = o;
}

// ---------------- bf16 GEMM: D = A(MxK) @ Bt(NxK)^T ----------------
// EPI 0: relu + bf16 store   EPI 1: fp32 store
template <int EPI>
__global__ __launch_bounds__(256)
void gemm_bf16(const bf16_t* __restrict__ A, const bf16_t* __restrict__ Bt,
               void* __restrict__ Dv, int M, int N, int K) {
  __shared__ bf16_t As[2][BM * BKP];
  __shared__ bf16_t Bs[2][BN * BKP];

  const int e = blockIdx.z;
  A  += (size_t)e * M * K;
  Bt += (size_t)e * N * K;

  const int m0   = blockIdx.y * BM;
  const int n0   = blockIdx.x * BN;
  const int tid  = threadIdx.x;
  const int lane = tid & 31;
  const int lr   = lane & 15;
  const int lh   = lane >> 4;
  const int wave = tid >> 5;
  const int wm   = (wave >> 2) * 64;  // 2 wave-rows
  const int wn   = (wave & 3)  * 64;  // 4 wave-cols

  // stage tile t into buffer b: 4 (A) + 8 (B) b128 copies per thread -> 12 async/wave
  auto stage = [&](int t, int b) {
    const int k0 = t * BK;
    #pragma unroll
    for (int i = 0; i < 4; ++i) {
      int v = tid + i * 256;
      int r = v >> 3;            // 8 chunks of 8 bf16 per 64-wide row
      int c = (v & 7) * 8;
      cp16(A + (size_t)(m0 + r) * K + k0 + c, &As[b][r * BKP + c]);
    }
    #pragma unroll
    for (int i = 0; i < 8; ++i) {
      int v = tid + i * 256;
      int r = v >> 3;
      int c = (v & 7) * 8;
      cp16(Bt + (size_t)(n0 + r) * K + k0 + c, &Bs[b][r * BKP + c]);
    }
  };

  v8f acc[4][4];
  #pragma unroll
  for (int i = 0; i < 4; ++i)
    #pragma unroll
    for (int j = 0; j < 4; ++j)
      #pragma unroll
      for (int q = 0; q < 8; ++q) acc[i][j][q] = 0.0f;

  // one K-stage of compute on buffer b: 32 wmma, 32 ds_load_b128
  auto compute = [&](int b) {
    const bf16_t* As_ = &As[b][0];
    const bf16_t* Bs_ = &Bs[b][0];
    #pragma unroll
    for (int ks = 0; ks < BK; ks += 32) {
      v16bf af[4];
      #pragma unroll
      for (int mi = 0; mi < 4; ++mi)
        af[mi] = frag_ld(As_, wm + mi * 16 + lr, ks + lh * 8, ks + 16 + lh * 8);
      #pragma unroll
      for (int ni = 0; ni < 4; ++ni) {
        v16bf bfr = frag_ld(Bs_, wn + ni * 16 + lr, ks + lh * 8, ks + 16 + lh * 8);
        #pragma unroll
        for (int mi = 0; mi < 4; ++mi)
          acc[mi][ni] = __builtin_amdgcn_wmma_f32_16x16x32_bf16(
              false, af[mi], false, bfr, (short)0, acc[mi][ni], false, false);
      }
    }
  };

  const int nT = K / BK;
  stage(0, 0);

  // steady state: issue next tile's DMA, wait for current tile (12 in flight)
  for (int t = 0; t < nT - 1; ++t) {
    stage(t + 1, (t + 1) & 1);
    WAIT_ASYNC(12);
    __syncthreads();
    compute(t & 1);
    __syncthreads();
  }
  WAIT_ASYNC(0);
  __syncthreads();
  compute((nT - 1) & 1);

  // Epilogue. D layout: lane = N, vgpr index = M within each 16x16 tile.
  #pragma unroll
  for (int mi = 0; mi < 4; ++mi)
    #pragma unroll
    for (int ni = 0; ni < 4; ++ni) {
      int col  = n0 + wn + ni * 16 + lr;
      int rowb = m0 + wm + mi * 16 + lh * 8;
      if (EPI == 0) {
        bf16_t* D = (bf16_t*)Dv + (size_t)e * M * N;
        #pragma unroll
        for (int v = 0; v < 8; ++v) {
          float x = acc[mi][ni][v];
          x = x > 0.0f ? x : 0.0f;
          D[(size_t)(rowb + v) * N + col] = (bf16_t)x;
        }
      } else {
        float* D = (float*)Dv + (size_t)e * M * N;
        #pragma unroll
        for (int v = 0; v < 8; ++v)
          D[(size_t)(rowb + v) * N + col] = acc[mi][ni][v];
      }
    }
}

extern "C" void kernel_launch(void* const* d_in, const int* in_sizes, int n_in,
                              void* d_out, int out_size, void* d_ws, size_t ws_size,
                              hipStream_t stream) {
  constexpr int E = 8, C = 2048, M = 2048, H = 8192;
  const float* x  = (const float*)d_in[0];   // (E, C, M)
  const float* w1 = (const float*)d_in[1];   // (E, M, H)
  const float* w2 = (const float*)d_in[2];   // (E, H, M)
  float* out = (float*)d_out;                // (E, C, M)

  // Workspace layout (bf16 elements)
  const size_t nX  = (size_t)E * C * M;      //  33.5M ->  64 MiB
  const size_t nW1 = (size_t)E * M * H;      // 134.2M -> 256 MiB
  const size_t nW2 = (size_t)E * H * M;      // 134.2M -> 256 MiB
  bf16_t* xb    = (bf16_t*)d_ws;             // (E, C, M)
  bf16_t* w1t   = xb + nX;                   // (E, H, M)  = W1^T per expert
  bf16_t* w2t   = w1t + nW1;                 // (E, M, H)  = W2^T per expert
  bf16_t* inter = w2t + nW2;                 // (E, C, H)

  // Pass 0: convert / transpose to bf16
  cast_f32_bf16<<<2048, 256, 0, stream>>>(x, xb, nX);
  {
    dim3 g(H / 32, M / 32, E);               // W1 (M x H) -> (H x M)
    transpose_cast_f32_bf16<<<g, 256, 0, stream>>>(w1, w1t, M, H);
  }
  {
    dim3 g(M / 32, H / 32, E);               // W2 (H x M) -> (M x H)
    transpose_cast_f32_bf16<<<g, 256, 0, stream>>>(w2, w2t, H, M);
  }

  // Pass 1: inter = relu(x @ W1)   A:(C x M)  Bt:(H x M)
  {
    dim3 g(H / BN, C / BM, E);
    gemm_bf16<0><<<g, 256, 0, stream>>>(xb, w1t, inter, C, H, M);
  }
  // Pass 2: out = inter @ W2       A:(C x H)  Bt:(M x H)
  {
    dim3 g(M / BN, C / BM, E);
    gemm_bf16<1><<<g, 256, 0, stream>>>(inter, w2t, out, C, M, H);
  }
}
